// ChebNet_CNN_DASE_76931454206315
// MI455X (gfx1250) — matrix operand order repaired and checked by reading, not compiled, not run on hardware
//
#include <hip/hip_runtime.h>
#include <hip/hip_bf16.h>

// ---------------------------------------------------------------------------
// MI455X (gfx1250) implementation. wave32, WGP, WMMA 16x16x32 bf16 (fp32 acc).
// Dominant cost: ~775 GFLOP of dense GEMM -> v_wmma_f32_16x16x32_bf16 path.
// GEMM staging is register-double-buffered with fast (unguarded) interior
// paths so global loads batch under one s_wait and overlap the WMMA pipe.
// ---------------------------------------------------------------------------

typedef __attribute__((ext_vector_type(16))) __bf16 v16bf;
typedef __attribute__((ext_vector_type(2)))  __bf16 bf16x2;
typedef __attribute__((ext_vector_type(8)))  float  v8f;

#define TM 128            // block tile M
#define TN 64             // block tile N
#define TK 32             // K step == WMMA K for bf16
#define AS 34             // LDS row stride for A tile (pad vs bank conflicts)
#define BS 34             // LDS row stride for B tile (stored transposed [n][k])

// ---- WMMA fragment builders (per CDNA5 ISA 7.12.2 16-bit layouts) ----------
// A 16x32: lanes 0-15 -> M=lane, K in {0..7,16..23}; lanes 16-31 -> same M,
// K in {8..15,24..31}; 2 bf16 per VGPR, low half first.
__device__ __forceinline__ v16bf frag_a(const __bf16* As_, int m0, int lane) {
  const __bf16* row = As_ + (m0 + (lane & 15)) * AS;
  const int kh = ((lane >> 4) & 1) << 3;  // 0 or 8
  v16bf a;
#pragma unroll
  for (int i = 0; i < 8; ++i) {
    const int kb = ((i >> 2) << 4) + kh + ((i & 3) << 1);
    a[2 * i]     = row[kb];
    a[2 * i + 1] = row[kb + 1];
  }
  return a;
}

// B 32x16: lanes 0-15 -> N=lane, K in {0..7,16..23}; lanes 16-31 -> N=lane-16, K in {8..15,24..31}.
// Bs is staged transposed ([n][k]) so each lane's K pairs are contiguous.
__device__ __forceinline__ v16bf frag_b(const __bf16* Bs_, int n0, int lane) {
  const __bf16* row = Bs_ + (n0 + (lane & 15)) * BS;
  const int kh = ((lane >> 4) & 1) << 3;  // 0 or 8
  v16bf b;
#pragma unroll
  for (int i = 0; i < 8; ++i) {
    const int kb = ((i >> 2) << 4) + kh + ((i & 3) << 1);
    b[2 * i]     = row[kb];
    b[2 * i + 1] = row[kb + 1];
  }
  return b;
}

// ---- Tiled WMMA GEMM: C = act(A*B [+C_old] [+bias]) ------------------------
// ACT: 0=none, 1=leaky_relu(0.01). ACCUM: add existing C. GATHER: A row i is
// emb[ri[i]][k] * emb[di[i]][k] (fused pairwise-product feature build).
template <int ACT, bool ACCUM, bool GATHER>
__global__ __launch_bounds__(256) void wmma_gemm_kernel(
    const float* __restrict__ A, int lda, const float* __restrict__ B, int ldb,
    const float* __restrict__ bias, float* __restrict__ C, int ldc,
    int M, int N, int K, const int* __restrict__ ri, const int* __restrict__ di) {
  __shared__ __bf16 As_[TM * AS];
  __shared__ __bf16 Bs_[TN * BS];
  __shared__ int rsel[TM], dsel[TM];

  const int tid  = threadIdx.x;
  const int lane = tid & 31;
  const int wave = tid >> 5;
  const int wm = (wave >> 1) * 32;  // 0,32,64,96
  const int wn = (wave & 1) * 32;   // 0,32
  const int bm = blockIdx.x * TM;
  const int bn = blockIdx.y * TN;

  if (GATHER) {
    if (tid < TM) {
      const int gr = bm + tid;
      rsel[tid] = (gr < M) ? ri[gr] : 0;
      dsel[tid] = (gr < M) ? di[gr] : 0;
    }
    __syncthreads();
  }

  const bool rows_in = (bm + TM) <= M;
  const bool cols_in = (bn + TN) <= N;

  // Per-thread ownership of the staging tiles:
  //   A: 8 k-pairs  -> idx2 = p*256+tid; r = idx2>>4;  kk = (idx2&15)*2
  //   B: 4 k-pairs  -> idx2 = p*256+tid; n = idx2&63;  kk = (idx2>>6)*2
  float ra[16], rb[8], ra2[16], rb2[8];

  auto load_a = [&](int k0, float (&r_)[16]) {
    const bool fast = rows_in && ((k0 + TK) <= K);
    if (fast) {
#pragma unroll
      for (int p = 0; p < 8; ++p) {
        const int idx2 = p * 256 + tid;
        const int r = idx2 >> 4;
        const int gk = k0 + ((idx2 & 15) << 1);
        if (GATHER) {
          const float* p1 = A + (size_t)rsel[r] * lda + gk;
          const float* p2 = A + (size_t)dsel[r] * lda + gk;
          r_[2 * p]     = p1[0] * p2[0];
          r_[2 * p + 1] = p1[1] * p2[1];
        } else {
          const float* pa = A + (size_t)(bm + r) * lda + gk;
          r_[2 * p]     = pa[0];
          r_[2 * p + 1] = pa[1];
        }
      }
    } else {
#pragma unroll
      for (int p = 0; p < 8; ++p) {
        const int idx2 = p * 256 + tid;
        const int r = idx2 >> 4;
        const int gk = k0 + ((idx2 & 15) << 1);
#pragma unroll
        for (int q = 0; q < 2; ++q) {
          float v = 0.0f;
          if ((bm + r) < M && (gk + q) < K) {
            if (GATHER)
              v = A[(size_t)rsel[r] * lda + gk + q] * A[(size_t)dsel[r] * lda + gk + q];
            else
              v = A[(size_t)(bm + r) * lda + gk + q];
          }
          r_[2 * p + q] = v;
        }
      }
    }
  };

  auto load_b = [&](int k0, float (&r_)[8]) {
    const bool fast = cols_in && ((k0 + TK) <= K);
    if (fast) {
#pragma unroll
      for (int p = 0; p < 4; ++p) {
        const int idx2 = p * 256 + tid;
        const int n = idx2 & 63;
        const int gk = k0 + ((idx2 >> 6) << 1);
        r_[2 * p]     = B[(size_t)gk * ldb + bn + n];
        r_[2 * p + 1] = B[(size_t)(gk + 1) * ldb + bn + n];
      }
    } else {
#pragma unroll
      for (int p = 0; p < 4; ++p) {
        const int idx2 = p * 256 + tid;
        const int n = idx2 & 63;
        const int gk = k0 + ((idx2 >> 6) << 1);
#pragma unroll
        for (int q = 0; q < 2; ++q) {
          float v = 0.0f;
          if ((gk + q) < K && (bn + n) < N) v = B[(size_t)(gk + q) * ldb + bn + n];
          r_[2 * p + q] = v;
        }
      }
    }
  };

  auto store_tiles = [&](const float (&a_)[16], const float (&b_)[8]) {
#pragma unroll
    for (int p = 0; p < 8; ++p) {
      const int idx2 = p * 256 + tid;
      const int r = idx2 >> 4;
      const int kk = (idx2 & 15) << 1;
      bf16x2 pv;
      pv[0] = (__bf16)a_[2 * p];
      pv[1] = (__bf16)a_[2 * p + 1];
      *reinterpret_cast<bf16x2*>(&As_[r * AS + kk]) = pv;   // one ds_store_b32
    }
#pragma unroll
    for (int p = 0; p < 4; ++p) {
      const int idx2 = p * 256 + tid;
      const int n = idx2 & 63;
      const int kk = (idx2 >> 6) << 1;
      bf16x2 pv;
      pv[0] = (__bf16)b_[2 * p];
      pv[1] = (__bf16)b_[2 * p + 1];
      *reinterpret_cast<bf16x2*>(&Bs_[n * BS + kk]) = pv;
    }
  };

  v8f acc[2][2] = {};

  load_a(0, ra);
  load_b(0, rb);

  for (int k0 = 0; k0 < K; k0 += TK) {
    store_tiles(ra, rb);
    __syncthreads();

    // Prefetch next tile into registers; its s_wait lands after the WMMAs.
    const int kn = k0 + TK;
    if (kn < K) {
      load_a(kn, ra2);
      load_b(kn, rb2);
    }

    const v16bf a0 = frag_a(As_, wm, lane);
    const v16bf a1 = frag_a(As_, wm + 16, lane);
    const v16bf b0 = frag_b(Bs_, wn, lane);
    const v16bf b1 = frag_b(Bs_, wn + 16, lane);
    acc[0][0] = __builtin_amdgcn_wmma_f32_16x16x32_bf16(false, a0, false, b0,
                                                        (short)0, acc[0][0], false, false);
    acc[0][1] = __builtin_amdgcn_wmma_f32_16x16x32_bf16(false, a0, false, b1,
                                                        (short)0, acc[0][1], false, false);
    acc[1][0] = __builtin_amdgcn_wmma_f32_16x16x32_bf16(false, a1, false, b0,
                                                        (short)0, acc[1][0], false, false);
    acc[1][1] = __builtin_amdgcn_wmma_f32_16x16x32_bf16(false, a1, false, b1,
                                                        (short)0, acc[1][1], false, false);
    __syncthreads();

    if (kn < K) {
#pragma unroll
      for (int i = 0; i < 16; ++i) ra[i] = ra2[i];
#pragma unroll
      for (int i = 0; i < 8; ++i) rb[i] = rb2[i];
    }
  }

  // Epilogue per C/D layout: lane<16 -> N=lane, M=vgpr; lane>=16 -> M=vgpr+8.
  const int half = lane >> 4;
  const int nl = lane & 15;
#pragma unroll
  for (int i = 0; i < 2; ++i) {
#pragma unroll
    for (int j = 0; j < 2; ++j) {
      const int col = bn + wn + j * 16 + nl;
      if (col < N) {
#pragma unroll
        for (int r = 0; r < 8; ++r) {
          const int row = bm + wm + i * 16 + half * 8 + r;
          if (row < M) {
            float v = acc[i][j][r];
            if (ACCUM) v += C[(size_t)row * ldc + col];
            if (bias) v += bias[col];
            if (ACT == 1) v = (v > 0.0f) ? v : 0.01f * v;
            C[(size_t)row * ldc + col] = v;
          }
        }
      }
    }
  }
}

// ---- Graph / elementwise helpers -------------------------------------------
__global__ void scale_copy_kernel(float* __restrict__ dst,
                                  const float* __restrict__ src, float beta,
                                  long n) {
  long i = (long)blockIdx.x * blockDim.x + threadIdx.x;
  if (i < n) dst[i] = src ? beta * src[i] : 0.0f;
}

__global__ void deg_kernel(float* __restrict__ deg, const int* __restrict__ ei,
                           int E) {
  int e = blockIdx.x * blockDim.x + threadIdx.x;
  if (e < E) atomicAdd(&deg[ei[E + e]], 1.0f);
}

__global__ void edgew_kernel(float* __restrict__ w, const float* __restrict__ deg,
                             const int* __restrict__ ei, int E) {
  int e = blockIdx.x * blockDim.x + threadIdx.x;
  if (e < E) {
    float ns = rsqrtf(fmaxf(deg[ei[e]], 1.0f));
    float nd = rsqrtf(fmaxf(deg[ei[E + e]], 1.0f));
    w[e] = ns * nd;
  }
}

// out[dst] += alpha * w[e] * h[src], 4 columns per thread (D=256 -> 64 thr/edge)
__global__ void scatter_kernel(float* __restrict__ out, const float* __restrict__ h,
                               const float* __restrict__ w, const int* __restrict__ ei,
                               float alpha, int E) {
  int gid = blockIdx.x * blockDim.x + threadIdx.x;
  if (gid >= E * 64) return;
  const int e = gid >> 6;
  const int c = (gid & 63) << 2;
  const int s = ei[e], d = ei[E + e];
  const float a = alpha * w[e];
  const float* hs = h + (size_t)s * 256 + c;
  float* od = out + (size_t)d * 256 + c;
#pragma unroll
  for (int q = 0; q < 4; ++q) atomicAdd(od + q, a * hs[q]);
}

// CNN branch: 4 1-D "VALID" convs (k=1,4,16,32; 6 ch) over each 256-wide row.
__global__ __launch_bounds__(256) void cnn_kernel(
    const float* __restrict__ res, const float* __restrict__ w1,
    const float* __restrict__ b1, const float* __restrict__ w4,
    const float* __restrict__ b4, const float* __restrict__ w16,
    const float* __restrict__ b16, const float* __restrict__ w32,
    const float* __restrict__ b32, float* __restrict__ out) {
  __shared__ float xs[256];
  const int n = blockIdx.x;
  xs[threadIdx.x] = res[(size_t)n * 256 + threadIdx.x];
  __syncthreads();
  float* o = out + (size_t)n * 5850;
  for (int j = threadIdx.x; j < 5850; j += 256) {
    float v;
    if (j < 1536) {                       // k=1, W=256
      int c = j >> 8, p = j & 255;
      v = xs[p] * w1[c] + b1[c];
    } else if (j < 3054) {                // k=4, W=253
      int jj = j - 1536, c = jj / 253, p = jj % 253;
      float s = b4[c];
#pragma unroll
      for (int i = 0; i < 4; ++i) s += xs[p + i] * w4[c * 4 + i];
      v = s;
    } else if (j < 4500) {                // k=16, W=241
      int jj = j - 3054, c = jj / 241, p = jj % 241;
      float s = b16[c];
#pragma unroll
      for (int i = 0; i < 16; ++i) s += xs[p + i] * w16[c * 16 + i];
      v = s;
    } else {                              // k=32, W=225
      int jj = j - 4500, c = jj / 225, p = jj % 225;
      float s = b32[c];
#pragma unroll
      for (int i = 0; i < 32; ++i) s += xs[p + i] * w32[c * 32 + i];
      v = s;
    }
    o[j] = fmaxf(v, 0.0f);
  }
}

__global__ void idx_kernel(int* __restrict__ ridx, int* __restrict__ didx,
                           const int* __restrict__ pr, const int* __restrict__ pd,
                           const int* __restrict__ nr, const int* __restrict__ nd,
                           int P, int NC) {
  int i = blockIdx.x * blockDim.x + threadIdx.x;
  if (i < 2 * P) {
    ridx[i] = (i < P) ? pr[i] : nr[i - P];
    didx[i] = NC + ((i < P) ? pd[i] : nd[i - P]);
  }
}

__global__ void label_kernel(float* __restrict__ out, int P) {
  int i = blockIdx.x * blockDim.x + threadIdx.x;
  if (i < 2 * P) out[i] = (i < P) ? 1.0f : 0.0f;
}

// Final 975->1 layer: one row per wave32, shuffle reduce, sigmoid.
__global__ __launch_bounds__(256) void mlp_final_kernel(
    const float* __restrict__ h3, const float* __restrict__ mw4,
    float* __restrict__ out, int M, int K) {
  const int wv = (blockIdx.x * blockDim.x + threadIdx.x) >> 5;
  const int lane = threadIdx.x & 31;
  if (wv >= M) return;
  const float* row = h3 + (size_t)wv * K;
  float s = 0.0f;
  for (int k = lane; k < K; k += 32) s += row[k] * mw4[k];
#pragma unroll
  for (int off = 16; off > 0; off >>= 1) s += __shfl_down(s, off, 32);
  if (lane == 0) out[wv] = 1.0f / (1.0f + __expf(-s));
}

// ---- Host-side dispatch -----------------------------------------------------
template <int ACT, bool ACCUM, bool GATHER>
static void gemm(hipStream_t st, const float* A, int lda, const float* B, int ldb,
                 const float* bias, float* C, int ldc, int M, int N, int K,
                 const int* ri = nullptr, const int* di = nullptr) {
  dim3 g((M + TM - 1) / TM, (N + TN - 1) / TN);
  wmma_gemm_kernel<ACT, ACCUM, GATHER><<<g, 256, 0, st>>>(A, lda, B, ldb, bias, C,
                                                          ldc, M, N, K, ri, di);
}

extern "C" void kernel_launch(void* const* d_in, const int* in_sizes, int n_in,
                              void* d_out, int out_size, void* d_ws, size_t ws_size,
                              hipStream_t stream) {
  (void)in_sizes; (void)n_in; (void)out_size; (void)ws_size;
  constexpr int NC = 8000, ND = 800, NN = 8800, FC = 8000, FD = 800, D = 256;
  constexpr int E = 281600, P = 8000, EMB = 5850, H1 = 2925, H2 = 1462, H3 = 975;
  constexpr int MP = 2 * P;  // 16000 training pairs

  const float* circ   = (const float*)d_in[0];
  const float* dis    = (const float*)d_in[1];
  const int*   ei     = (const int*)d_in[2];
  const int*   pos_r  = (const int*)d_in[3];
  const int*   pos_d  = (const int*)d_in[4];
  const int*   neg_r  = (const int*)d_in[5];
  const int*   neg_d  = (const int*)d_in[6];
  const float* Wc     = (const float*)d_in[7];
  const float* Wd     = (const float*)d_in[8];
  const float* chebW  = (const float*)d_in[9];
  const float* chebB  = (const float*)d_in[10];
  const float* cw1    = (const float*)d_in[11];
  const float* cb1    = (const float*)d_in[12];
  const float* cw4    = (const float*)d_in[13];
  const float* cb4    = (const float*)d_in[14];
  const float* cw16   = (const float*)d_in[15];
  const float* cb16   = (const float*)d_in[16];
  const float* cw32   = (const float*)d_in[17];
  const float* cb32   = (const float*)d_in[18];
  const float* mw1    = (const float*)d_in[19];
  const float* mb1    = (const float*)d_in[20];
  const float* mw2    = (const float*)d_in[21];
  const float* mb2    = (const float*)d_in[22];
  const float* mw3    = (const float*)d_in[23];
  const float* mb3    = (const float*)d_in[24];
  const float* mw4    = (const float*)d_in[25];

  float* outp = (float*)d_out;           // [score 16000 | label 16000 | cnn 8800*5850]
  float* cnn_out = outp + 2 * MP;

  // Workspace carve (~327 MB total): 4 rotating Cheb buffers, res, deg,
  // edge weights, gather indices, two MLP activations (h3 reuses h1's slot).
  float* wsf = (float*)d_ws;
  size_t off = 0;
  auto carve = [&](size_t n) { float* p = wsf + off; off += (n + 255) & ~(size_t)255; return p; };
  float* X0    = carve((size_t)NN * D);
  float* X1b   = carve((size_t)NN * D);
  float* X2b   = carve((size_t)NN * D);
  float* X3b   = carve((size_t)NN * D);
  float* res   = carve((size_t)NN * D);
  float* deg   = carve(NN);
  float* wedge = carve(E);
  int*   ridx  = (int*)carve(MP);
  int*   didx  = (int*)carve(MP);
  float* h1    = carve((size_t)MP * H1);
  float* h2    = carve((size_t)MP * H2);
  float* h3    = h1;  // h1 dead once h2 exists; 16000*975 fits in its slot

  const long nnD = (long)NN * D;
  const int scat_blocks = (E * 64 + 255) / 256;

  // 1) h = [circ @ Wc ; dis @ Wd]   (fp32->bf16 WMMA, fp32 accumulate)
  gemm<0, false, false>(stream, circ, FC, Wc, D, nullptr, X0, D, NC, D, FC);
  gemm<0, false, false>(stream, dis, FD, Wd, D, nullptr, X0 + (size_t)NC * D, D, ND, D, FD);

  // 2) Edge normalization w = deg^-1/2[src] * deg^-1/2[dst]
  scale_copy_kernel<<<(NN + 255) / 256, 256, 0, stream>>>(deg, nullptr, 0.0f, NN);
  deg_kernel<<<(E + 255) / 256, 256, 0, stream>>>(deg, ei, E);
  edgew_kernel<<<(E + 255) / 256, 256, 0, stream>>>(wedge, deg, ei, E);

  // 3) Chebyshev recurrence: res = sum_k X_k @ W_k (+ bias on last term)
  gemm<0, false, false>(stream, X0, D, chebW, D, nullptr, res, D, NN, D, D);
  scale_copy_kernel<<<(nnD + 255) / 256, 256, 0, stream>>>(X1b, nullptr, 0.0f, nnD);
  scatter_kernel<<<scat_blocks, 256, 0, stream>>>(X1b, X0, wedge, ei, -1.0f, E);
  gemm<0, true, false>(stream, X1b, D, chebW + (size_t)1 * D * D, D, nullptr, res, D, NN, D, D);

  float* bufs[4] = {X0, X1b, X2b, X3b};
  float* Xpp = X0;
  float* Xp = X1b;
  for (int k = 2; k < 6; ++k) {
    float* Xk = bufs[k & 3];  // k=2:X2, 3:X3, 4:X0(reuse), 5:X1(reuse)
    scale_copy_kernel<<<(nnD + 255) / 256, 256, 0, stream>>>(Xk, Xpp, -1.0f, nnD);
    scatter_kernel<<<scat_blocks, 256, 0, stream>>>(Xk, Xp, wedge, ei, -2.0f, E);
    gemm<0, true, false>(stream, Xk, D, chebW + (size_t)k * D * D, D,
                         (k == 5) ? chebB : nullptr, res, D, NN, D, D);
    Xpp = Xp; Xp = Xk;
  }

  // 4) CNN branch -> cnn_outputs straight into d_out
  cnn_kernel<<<NN, 256, 0, stream>>>(res, cw1, cb1, cw4, cb4, cw16, cb16, cw32,
                                     cb32, cnn_out);

  // 5) Labels + gather indices
  label_kernel<<<(MP + 255) / 256, 256, 0, stream>>>(outp + MP, P);
  idx_kernel<<<(MP + 255) / 256, 256, 0, stream>>>(ridx, didx, pos_r, pos_d,
                                                   neg_r, neg_d, P, NC);

  // 6) MLP: layer1 fuses train_feats = emb_circ[r] * emb_dis[d] into the
  //    GEMM A-stage (GATHER mode) -- avoids a 374 MB intermediate.
  gemm<1, false, true>(stream, cnn_out, EMB, mw1, H1, mb1, h1, H1, MP, H1, EMB, ridx, didx);
  gemm<1, false, false>(stream, h1, H1, mw2, H2, mb2, h2, H2, MP, H2, H1);
  gemm<1, false, false>(stream, h2, H2, mw3, H3, mb3, h3, H3, MP, H3, H2);
  mlp_final_kernel<<<(MP * 32 + 255) / 256, 256, 0, stream>>>(h3, mw4, outp, MP, H3);
}
